// CausalSelfAttention_7103875907933
// MI455X (gfx1250) — matrix operand order repaired and checked by reading, not compiled, not run on hardware
//
#include <hip/hip_runtime.h>

// ---------------------------------------------------------------------------
// CDNA5 (gfx1250) causal self-attention, wave32 + v_wmma_f32_16x16x32_bf16.
// ---------------------------------------------------------------------------

typedef __attribute__((ext_vector_type(16))) __bf16 v16bf;
typedef __attribute__((ext_vector_type(8)))  float  v8f;

union AB16 { uint4 q[2]; v16bf v; };
union V8U  { v8f v; float f[8]; };

// v_exp_f32 is base-2 natively
#define EXP2F(x) __builtin_amdgcn_exp2f(x)

// fp32 -> bf16 round-to-nearest-even (bit trick; no scalar bf16 arithmetic)
__device__ __forceinline__ unsigned short f2bf(float f) {
  unsigned int u = __float_as_uint(f);
  u += 0x7FFFu + ((u >> 16) & 1u);
  return (unsigned short)(u >> 16);
}

// A-fragment (16x32 bf16, M x K): lane m = lane%16; half h = lane/16 holds
// K = h*8..h*8+7 (VGPR0-3) and K = 16+h*8..+7 (VGPR4-7). Two 16B loads.
__device__ __forceinline__ v16bf load_a_frag(const unsigned short* base, int ld,
                                             int m0, int k0, int lane) {
  const unsigned short* p =
      base + (size_t)(m0 + (lane & 15)) * ld + k0 + (lane >> 4) * 8;
  AB16 u;
  u.q[0] = *reinterpret_cast<const uint4*>(p);
  u.q[1] = *reinterpret_cast<const uint4*>(p + 16);
  return u.v;
}

// B-fragment (32x16 bf16, K x N) sourced from a row-major [N,K] operand
// (computing A*B^T): lane n = lane%16, half h holds K = h*8..h*8+7
// (VGPR0-3) and K = 16+h*8..+7 (VGPR4-7). Two 16B loads.
__device__ __forceinline__ v16bf load_b_frag(const unsigned short* base, int ld,
                                             int n0, int k0, int lane) {
  const unsigned short* p =
      base + (size_t)(n0 + (lane & 15)) * ld + k0 + (lane >> 4) * 8;
  AB16 u;
  u.q[0] = *reinterpret_cast<const uint4*>(p);
  u.q[1] = *reinterpret_cast<const uint4*>(p + 16);
  return u.v;
}

// ---------------------------------------------------------------------------
// fp32 -> bf16 bulk convert (vectorized x4)
// ---------------------------------------------------------------------------
__global__ void f32_to_bf16_kernel(const float4* __restrict__ in,
                                   ushort4* __restrict__ out, int n4) {
  int i = blockIdx.x * blockDim.x + threadIdx.x;
  if (i >= n4) return;
  float4 f = in[i];
  ushort4 o;
  o.x = f2bf(f.x); o.y = f2bf(f.y); o.z = f2bf(f.z); o.w = f2bf(f.w);
  out[i] = o;
}

// ---------------------------------------------------------------------------
// V transpose with interleaved K-position order.
// Within each 32-wide t-block, position p holds t = base | ((p&1)<<4) | (p>>1).
// This matches the packed-u32 LDS layout of exp(P) in the attention kernel
// (P*V is invariant under a shared permutation of the K axis).
// ---------------------------------------------------------------------------
__global__ void transpose_v_kernel(const unsigned short* __restrict__ qkv,
                                   unsigned short* __restrict__ vt) {
  constexpr int T = 2048, D = 1024, LDQ = 3 * D;
  size_t i = (size_t)blockIdx.x * blockDim.x + threadIdx.x;  // B*H*64*T
  int pos = (int)(i & (T - 1));
  int p = pos & 31;
  int t = (pos & ~31) | ((p & 1) << 4) | (p >> 1);
  size_t r = i >> 11;
  int dk = (int)(r & 63);
  int hd = (int)((r >> 6) & 15);
  int b  = (int)(r >> 10);
  vt[i] = qkv[(size_t)(b * T + t) * LDQ + 2 * D + hd * 64 + dk];
}

// ---------------------------------------------------------------------------
// Register-blocked bf16 GEMM: C[M,N] = A[M,K] * B[N,K]^T, fp32 accumulate.
// 64x64 per wave (4x4 WMMA tiles), 8 waves/block, no LDS (operands stream
// through the 192MB L2, which holds them entirely).
// ---------------------------------------------------------------------------
template <bool OUT_BF16>
__global__ void __launch_bounds__(256)
gemm_bf16_abt_kernel(const unsigned short* __restrict__ A,
                     const unsigned short* __restrict__ Bm,
                     void* __restrict__ Cout, int M, int N, int K) {
  const int lane = threadIdx.x & 31;
  const int wid  = blockIdx.x * (blockDim.x >> 5) + (threadIdx.x >> 5);
  const int tn   = N >> 6;
  const int wm   = wid / tn, wn = wid % tn;
  if (wm * 64 >= M) return;
  const int m0 = wm * 64, n0 = wn * 64;

  v8f acc[4][4] = {};
  for (int k = 0; k < K; k += 32) {
    v16bf a[4], b[4];
#pragma unroll
    for (int i = 0; i < 4; ++i) a[i] = load_a_frag(A, K, m0 + 16 * i, k, lane);
#pragma unroll
    for (int j = 0; j < 4; ++j) b[j] = load_b_frag(Bm, K, n0 + 16 * j, k, lane);
#pragma unroll
    for (int i = 0; i < 4; ++i)
#pragma unroll
      for (int j = 0; j < 4; ++j)
        acc[i][j] = __builtin_amdgcn_wmma_f32_16x16x32_bf16(
            false, a[i], false, b[j], (short)0, acc[i][j], false, false);
  }

  const int ln = lane & 15, lh = lane >> 4;
#pragma unroll
  for (int i = 0; i < 4; ++i)
#pragma unroll
    for (int j = 0; j < 4; ++j) {
      V8U u; u.v = acc[i][j];
      int n = n0 + 16 * j + ln;
#pragma unroll
      for (int r = 0; r < 8; ++r) {
        size_t idx = (size_t)(m0 + 16 * i + r + 8 * lh) * N + n;
        if (OUT_BF16)
          reinterpret_cast<unsigned short*>(Cout)[idx] = f2bf(u.f[r]);
        else
          reinterpret_cast<float*>(Cout)[idx] = u.f[r];
      }
    }
}

// ---------------------------------------------------------------------------
// Flash attention building block: process G*32 key columns starting at kt.
//   acc[0..3] : O accumulator (16 x 64, fp32)
//   acc[4]    : row-sum accumulator via ones-WMMA (replaces shuffle-add + li)
// Row max uses a width-16 butterfly; exp uses base-2 with the 1/sqrt(dk)
// scale folded into log2(e)/8. exp(P) is packed 2-per-dword into LDS with an
// interleaved position order matched by the permuted V^T layout.
// ---------------------------------------------------------------------------
#define ROWU 72  // ushorts per LDS P-row: 64 positions + 8 pad (144B rows: 16B-aligned, bank-split halves)

template <int G, bool MASKED>
__device__ __forceinline__ void attn_step(
    int kt, int q0, int lane,
    const unsigned short* __restrict__ kbase,   // K section, head column pre-folded
    const unsigned short* __restrict__ vbase,   // V^T [dk, T] (position-permuted)
    v16bf qa0, v16bf qa1, V8U* acc, float* mi,
    unsigned short* psh)                        // this wave's LDS slab
{
  constexpr int LDQ = 3 * 1024, T = 2048;
  const int ln = lane & 15, lh = lane >> 4;

  // S = Q K^T for 2G score tiles of 16 columns
  V8U s[2 * G];
#pragma unroll
  for (int f = 0; f < 2 * G; ++f) {
    v8f sv = {};
    v16bf kb = load_b_frag(kbase, LDQ, kt + f * 16, 0, lane);
    sv = __builtin_amdgcn_wmma_f32_16x16x32_bf16(false, qa0, false, kb, (short)0, sv, false, false);
    kb = load_b_frag(kbase, LDQ, kt + f * 16, 32, lane);
    sv = __builtin_amdgcn_wmma_f32_16x16x32_bf16(false, qa1, false, kb, (short)0, sv, false, false);
    s[f].v = sv;
  }

  const float c1 = 0.18033688011112042f;  // log2(e) / sqrt(64)
#pragma unroll
  for (int r = 0; r < 8; ++r) {
    float e[2 * G];
    float mx = -3.0e38f;
    const int mg = q0 + r + 8 * lh;
#pragma unroll
    for (int f = 0; f < 2 * G; ++f) {
      float v = s[f].f[r] * c1;
      if (MASKED && (kt + f * 16 + ln > mg)) v = -3.0e38f;
      e[f] = v;
      mx = fmaxf(mx, v);
    }
    for (int off = 1; off < 16; off <<= 1)
      mx = fmaxf(mx, __shfl_xor(mx, off, 32));
    const float mnew  = fmaxf(mi[r], mx);
    const float alpha = EXP2F(mi[r] - mnew);
    mi[r] = mnew;
#pragma unroll
    for (int a = 0; a < 5; ++a) acc[a].f[r] *= alpha;
    const int m = r + 8 * lh;
#pragma unroll
    for (int g = 0; g < G; ++g) {
      const float e0 = EXP2F(e[2 * g] - mnew);       // column g*32 + ln
      const float e1 = EXP2F(e[2 * g + 1] - mnew);   // column g*32 + 16 + ln
      const unsigned int pk = ((unsigned int)f2bf(e1) << 16) | f2bf(e0);
      *reinterpret_cast<unsigned int*>(&psh[m * ROWU + g * 32 + ln * 2]) = pk;
    }
  }

  // same-wave DS ordering before the transposed readback
  asm volatile("s_wait_dscnt 0" ::: "memory");

  AB16 ones;
  ones.q[0] = make_uint4(0x3F803F80u, 0x3F803F80u, 0x3F803F80u, 0x3F803F80u);
  ones.q[1] = ones.q[0];

#pragma unroll
  for (int g = 0; g < G; ++g) {
    AB16 up;
    const unsigned short* pr = &psh[ln * ROWU + g * 32 + lh * 8];
    up.q[0] = *reinterpret_cast<const uint4*>(pr);
    up.q[1] = *reinterpret_cast<const uint4*>(pr + 16);
    const v16bf pa = up.v;
#pragma unroll
    for (int ni = 0; ni < 4; ++ni) {
      v16bf vb = load_b_frag(vbase, T, ni * 16, kt + g * 32, lane);
      acc[ni].v = __builtin_amdgcn_wmma_f32_16x16x32_bf16(
          false, pa, false, vb, (short)0, acc[ni].v, false, false);
    }
    // row-sum of exp(P): one WMMA against all-ones B replaces shuffle-adds
    acc[4].v = __builtin_amdgcn_wmma_f32_16x16x32_bf16(
        false, pa, false, ones.v, (short)0, acc[4].v, false, false);
  }
}

__global__ void __launch_bounds__(256)
flash_attn_kernel(const unsigned short* __restrict__ qkv,  // [B*T, 3D]
                  const unsigned short* __restrict__ vt,   // [B,H,DK,T] permuted
                  unsigned short* __restrict__ attn) {     // [B*T, D]
  constexpr int T = 2048, D = 1024, H = 16, DK = 64, LDQ = 3 * D;
  __shared__ unsigned short psh[8][16 * ROWU];  // 2.25KB per wave

  const int lane = threadIdx.x & 31;
  const int w    = threadIdx.x >> 5;
  const int wid  = blockIdx.x * 8 + w;
  const int qt = wid & (T / 16 - 1);
  const int hd = (wid >> 7) & (H - 1);
  const int b  = wid >> 11;
  const int q0 = qt * 16;
  const int ln = lane & 15, lh = lane >> 4;

  const unsigned short* qbase = qkv + (size_t)b * T * LDQ + hd * 64;
  const unsigned short* kbase = qbase + D;
  const unsigned short* vbase = vt + (size_t)((b * H + hd) * DK) * T;

  const v16bf qa0 = load_a_frag(qbase, LDQ, q0, 0,  lane);
  const v16bf qa1 = load_a_frag(qbase, LDQ, q0, 32, lane);

  V8U acc[5];
#pragma unroll
  for (int a = 0; a < 5; ++a)
#pragma unroll
    for (int r = 0; r < 8; ++r) acc[a].f[r] = 0.f;
  float mi[8];
#pragma unroll
  for (int r = 0; r < 8; ++r) mi[r] = -3.0e38f;

  // phase 1: full 64-wide blocks, branch-free
  const int nfull = q0 >> 6;
  for (int j = 0; j < nfull; ++j)
    attn_step<2, false>(j * 64, q0, lane, kbase, vbase, qa0, qa1, acc, mi, psh[w]);

  // phase 2: <=2 diagonal 32-wide blocks with causal mask
  const int j32end = (q0 + 15) >> 5;
  for (int j = nfull * 2; j <= j32end; ++j)
    attn_step<1, true>(j * 32, q0, lane, kbase, vbase, qa0, qa1, acc, mi, psh[w]);

  // epilogue: normalize by the ones-WMMA row sums, scatter bf16
#pragma unroll
  for (int r = 0; r < 8; ++r) {
    const float inv = 1.0f / acc[4].f[r];
    const int row = b * T + q0 + r + 8 * lh;
#pragma unroll
    for (int ni = 0; ni < 4; ++ni)
      attn[(size_t)row * D + hd * 64 + ni * 16 + ln] = f2bf(acc[ni].f[r] * inv);
  }
}

// ---------------------------------------------------------------------------
// Launcher
// ---------------------------------------------------------------------------
extern "C" void kernel_launch(void* const* d_in, const int* in_sizes, int n_in,
                              void* d_out, int out_size, void* d_ws, size_t ws_size,
                              hipStream_t stream) {
  constexpr int Bc = 4, Tc = 2048, Dc = 1024, Hc = 16;
  constexpr int BT = Bc * Tc;  // 8192

  const float* x     = (const float*)d_in[0];
  const float* w_qkv = (const float*)d_in[1];
  const float* w_o   = (const float*)d_in[2];
  float* out = (float*)d_out;

  // workspace layout (bf16 = unsigned short), ~104 MB total
  unsigned short* xb    = (unsigned short*)d_ws;
  unsigned short* wqkvb = xb    + (size_t)BT * Dc;
  unsigned short* wob   = wqkvb + (size_t)3 * Dc * Dc;
  unsigned short* qkv   = wob   + (size_t)Dc * Dc;
  unsigned short* vtb   = qkv   + (size_t)BT * 3 * Dc;
  unsigned short* attn  = vtb   + (size_t)BT * Dc;

  const int nx = BT * Dc;       // 8388608
  const int nq = 3 * Dc * Dc;   // 3145728
  const int no = Dc * Dc;       // 1048576

  f32_to_bf16_kernel<<<nx / 4 / 256, 256, 0, stream>>>((const float4*)x, (ushort4*)xb, nx / 4);
  f32_to_bf16_kernel<<<nq / 4 / 256, 256, 0, stream>>>((const float4*)w_qkv, (ushort4*)wqkvb, nq / 4);
  f32_to_bf16_kernel<<<no / 4 / 256, 256, 0, stream>>>((const float4*)w_o, (ushort4*)wob, no / 4);

  // QKV projection: [8192,1024] x [3072,1024]^T -> bf16 [8192,3072]
  gemm_bf16_abt_kernel<true><<<(BT / 64) * (3 * Dc / 64) / 8, 256, 0, stream>>>(
      xb, wqkvb, qkv, BT, 3 * Dc, Dc);

  // V -> [b,h,dk,T] with interleaved 32-block position order
  transpose_v_kernel<<<nx / 256, 256, 0, stream>>>(qkv, vtb);

  // Flash attention: 8192 wave-tiles, 8 waves/block
  flash_attn_kernel<<<(Bc * Hc * (Tc / 16)) / 8, 256, 0, stream>>>(qkv, vtb, attn);

  // Output projection: [8192,1024] x [1024,1024]^T -> fp32 d_out
  gemm_bf16_abt_kernel<false><<<(BT / 64) * (Dc / 64) / 8, 256, 0, stream>>>(
      attn, wob, out, BT, Dc, Dc);
}
